// RGB_SFMCNN_V3_40114994544664
// MI455X (gfx1250) — matrix operand order repaired and checked by reading, not compiled, not run on hardware
//
#include <hip/hip_runtime.h>

typedef __attribute__((ext_vector_type(16))) _Float16 v16h;
typedef __attribute__((ext_vector_type(8)))  _Float16 v8h;
typedef __attribute__((ext_vector_type(8)))  float    v8f;

#define BATCH 32
#define CIN   3
#define HH    224
#define WW    224
#define HP    228                 // padded H/W (2-halo each side)
#define OCH   64
#define DD    75                  // CIN*5*5
#define KPAD  96                  // 3 hardware K-slabs of 32
#define LPIX  (HH*WW)             // 50176
#define PPLANE (HP*HP)            // 51984
#define PIMG  (CIN*PPLANE)        // 155952
#define T32_PER_IMG (LPIX/32)     // 1568
#define WAVES_PER_BLK 8
#define WH_ELEMS (OCH*KPAD)       // 6144 halves
#define WS_WH_BYTES (WH_ELEMS*2)  // 12288
#define XP_OFF      (WS_WH_BYTES + OCH*4)          // 12544 (16B aligned)
#define WS_TOTAL    ((size_t)XP_OFF + (size_t)BATCH*PIMG*4)  // ~19.97 MB

// Logical kernel-row rho = c*5 + i (15 real rows; rho in [15,18) = padding rows).
__device__ constexpr int rowoffP(int rho) {        // padded-image row offset
    return (rho < 15) ? (((rho / 5) * HP + (rho % 5)) * HP) : 0;
}
__device__ constexpr int rowoff_u(int rho) {       // unpadded (fallback)
    return (rho < 15) ? (((rho / 5) * HH + (rho % 5) - 2) * WW) : 0;
}
__device__ constexpr int rowdy_u(int rho) {
    return (rho < 15) ? ((rho % 5) - 2) : 100000;
}
__device__ constexpr float rowmsk(int ks, int g_is1, int q) {
    return ((6 * ks + 3 * g_is1 + q) < 15) ? 1.0f : 0.0f;  // only ks==2,g==1 invalid
}

// ---- one-time: K-permuted f16 weights + ||w||^2 into workspace ----
__global__ __launch_bounds__(256)
void rbf_prep(const float* __restrict__ w, _Float16* __restrict__ whp,
              float* __restrict__ wnp)
{
    const int tid = threadIdx.x;
    for (int idx = tid; idx < WH_ELEMS; idx += 256) {
        int n   = idx / KPAD, t = idx - n * KPAD;
        int ks  = t >> 5, khw = t & 31;
        int hi  = khw >> 4, rem = khw & 15;
        int g   = hi, e = rem;
        float val = 0.0f;
        if (e < 15) {
            int q = e / 5, j = e - q * 5;
            int rho = 6 * ks + 3 * g + q;
            if (rho < 15) val = w[n * DD + rho * 5 + j];
        }
        whp[idx] = (_Float16)val;
    }
    if (tid < OCH) {
        float s = 0.0f;
        for (int k = 0; k < DD; ++k) { float v = w[tid * DD + k]; s += v * v; }
        wnp[tid] = s;
    }
}

// ---- one-time: zero-padded input copy (b,c,228,228) ----
__global__ __launch_bounds__(256)
void rbf_pad(const float* __restrict__ x, float* __restrict__ xp)
{
    int i = blockIdx.x * 256 + threadIdx.x;
    const int N = BATCH * PIMG;
    if (i >= N) return;
    int bc = i / PPLANE;
    int p  = i - bc * PPLANE;
    int yy = p / HP;
    int xx = p - yy * HP;
    int yu = yy - 2, xu = xx - 2;
    float v = 0.0f;
    if ((unsigned)yu < HH && (unsigned)xu < WW)
        v = x[(size_t)bc * LPIX + yu * WW + xu];
    xp[i] = v;
}

// ---- shared epilogue: distances for one 16-pixel half + coalesced stores ----
__device__ __forceinline__
void epilogue_half(float* dw, v8f a0, v8f a1, v8f a2, v8f a3,
                   float pnh, int g, int m, int lane,
                   const float* wn_s, float* out, int b, int lh)
{
    float pn_full = pnh + __shfl_xor(pnh, 16, 32);
    float pnm[8];
    #pragma unroll
    for (int r = 0; r < 8; ++r) pnm[r] = __shfl(pn_full, 8 * g + r, 32);

    #pragma unroll
    for (int nt = 0; nt < 4; ++nt) {
        v8f acc = (nt == 0) ? a0 : (nt == 1) ? a1 : (nt == 2) ? a2 : a3;
        int   n  = nt * 16 + m;
        float wn = wn_s[n];
        #pragma unroll
        for (int r = 0; r < 8; ++r) {
            float d2 = pnm[r] + wn - 2.0f * acc[r];
            dw[n * 17 + (r + 8 * g)] = sqrtf(fmaxf(d2, 0.0f));
        }
    }
    #pragma unroll
    for (int t = 0; t < 8; ++t) {
        int idx = t * 32 + lane;
        int n   = idx >> 2;
        int cm  = (idx & 3) * 4;
        float4 vv = make_float4(dw[n * 17 + cm + 0], dw[n * 17 + cm + 1],
                                dw[n * 17 + cm + 2], dw[n * 17 + cm + 3]);
        float4* op = (float4*)(out + ((size_t)(b * OCH + n)) * LPIX + lh + cm);
        *op = vv;
    }
}

// ================= main kernel, padded-input path (uniform, branch-free) ====
__global__ __launch_bounds__(256)
void rbf_main_padded(const float* __restrict__ xp,
                     const _Float16* __restrict__ whp,
                     const float* __restrict__ wnp,
                     float* __restrict__ out)
{
    __shared__ __align__(16) _Float16 Wh[WH_ELEMS];
    __shared__ float wn_s[OCH];
    __shared__ float dist_s[WAVES_PER_BLK][OCH * 17];

    const int tid = threadIdx.x;
    {   // bulk weight stage: 768 uint4 total
        const uint4* src = (const uint4*)whp;
        uint4*       dst = (uint4*)Wh;
        #pragma unroll
        for (int i = 0; i < 3; ++i) dst[tid + i * 256] = src[tid + i * 256];
        if (tid < OCH) wn_s[tid] = wnp[tid];
    }
    __syncthreads();

    const int wave = tid >> 5;
    const int lane = tid & 31;
    const int g    = lane >> 4;
    const int m    = lane & 15;

    const int tile = blockIdx.x * WAVES_PER_BLK + wave;  // < 50176
    const int b    = tile / T32_PER_IMG;
    const int lt   = tile - b * T32_PER_IMG;
    const int l0   = lt * 32;
    const int y    = l0 / WW;
    const int x0   = l0 - y * WW;
    const int xu   = x0 + m;                 // padded col of pixel (halo absorbs -2)

    const float* xpb = xp + (size_t)b * PIMG;
    const int    pb  = y * HP + xu;

    v8f c00 = {}, c01 = {}, c02 = {}, c03 = {};
    v8f c10 = {}, c11 = {}, c12 = {}, c13 = {};
    float pn0 = 0.0f, pn1 = 0.0f;

    #pragma unroll
    for (int ks = 0; ks < 3; ++ks) {
        v16h a0, a1;
        #pragma unroll
        for (int q = 0; q < 3; ++q) {
            int ro = g ? rowoffP(6 * ks + 3 + q) : rowoffP(6 * ks + q);
            float msk = g ? rowmsk(ks, 1, q) : 1.0f;     // folds to 1.0 except ks==2,g==1
            const float* p = xpb + (ro + pb);            // one addr, 10 imm-offset loads
            #pragma unroll
            for (int j = 0; j < 5; ++j) {
                float v0 = p[j]      * msk;
                float v1 = p[j + 16] * msk;
                pn0 += v0 * v0;  pn1 += v1 * v1;
                a0[q * 5 + j] = (_Float16)v0;
                a1[q * 5 + j] = (_Float16)v1;
            }
        }
        a0[15] = (_Float16)0.0f;
        a1[15] = (_Float16)0.0f;

        #pragma unroll
        for (int nt = 0; nt < 4; ++nt) {
            int n = nt * 16 + m;
            const v8h* bp = (const v8h*)&Wh[n * KPAD + ks * 32 + g * 16];
            v8h blo = bp[0], bhi = bp[1];
            v16h bf;
            #pragma unroll
            for (int i2 = 0; i2 < 8; ++i2) { bf[i2] = blo[i2]; bf[8 + i2] = bhi[i2]; }
            switch (nt) {
            case 0: c00 = __builtin_amdgcn_wmma_f32_16x16x32_f16(false, a0, false, bf, (short)0, c00, false, false);
                    c10 = __builtin_amdgcn_wmma_f32_16x16x32_f16(false, a1, false, bf, (short)0, c10, false, false); break;
            case 1: c01 = __builtin_amdgcn_wmma_f32_16x16x32_f16(false, a0, false, bf, (short)0, c01, false, false);
                    c11 = __builtin_amdgcn_wmma_f32_16x16x32_f16(false, a1, false, bf, (short)0, c11, false, false); break;
            case 2: c02 = __builtin_amdgcn_wmma_f32_16x16x32_f16(false, a0, false, bf, (short)0, c02, false, false);
                    c12 = __builtin_amdgcn_wmma_f32_16x16x32_f16(false, a1, false, bf, (short)0, c12, false, false); break;
            default:c03 = __builtin_amdgcn_wmma_f32_16x16x32_f16(false, a0, false, bf, (short)0, c03, false, false);
                    c13 = __builtin_amdgcn_wmma_f32_16x16x32_f16(false, a1, false, bf, (short)0, c13, false, false); break;
            }
        }
    }

    float* dw = dist_s[wave];
    epilogue_half(dw, c00, c01, c02, c03, pn0, g, m, lane, wn_s, out, b, l0);
    epilogue_half(dw, c10, c11, c12, c13, pn1, g, m, lane, wn_s, out, b, l0 + 16);
}

// ================= fallback (no workspace): in-kernel staging + bounds ======
__global__ __launch_bounds__(256)
void rbf_main_fallback(const float* __restrict__ x,
                       const float* __restrict__ w,
                       float* __restrict__ out)
{
    __shared__ __align__(16) _Float16 Wh[WH_ELEMS];
    __shared__ float wn_s[OCH];
    __shared__ float dist_s[WAVES_PER_BLK][OCH * 17];

    const int tid = threadIdx.x;
    for (int idx = tid; idx < WH_ELEMS; idx += 256) {
        int n   = idx / KPAD, t = idx - n * KPAD;
        int ks  = t >> 5, khw = t & 31;
        int hi  = khw >> 4, rem = khw & 15;
        int g   = hi, e = rem;
        float val = 0.0f;
        if (e < 15) {
            int q = e / 5, j = e - q * 5;
            int rho = 6 * ks + 3 * g + q;
            if (rho < 15) val = w[n * DD + rho * 5 + j];
        }
        Wh[idx] = (_Float16)val;
    }
    if (tid < OCH) {
        float s = 0.0f;
        for (int k = 0; k < DD; ++k) { float v = w[tid * DD + k]; s += v * v; }
        wn_s[tid] = s;
    }
    __syncthreads();

    const int wave = tid >> 5;
    const int lane = tid & 31;
    const int g    = lane >> 4;
    const int m    = lane & 15;

    const int tile = blockIdx.x * WAVES_PER_BLK + wave;
    const int b    = tile / T32_PER_IMG;
    const int lt   = tile - b * T32_PER_IMG;
    const int l0   = lt * 32;
    const int y    = l0 / WW;
    const int x0   = l0 - y * WW;
    const int xw   = x0 + m;

    const float* xb = x + (size_t)b * CIN * HH * WW;
    const int    pb = y * WW + xw - 2;
    const bool interior = (y >= 2) && (y <= HH - 3) && (x0 >= 2) && (x0 <= WW - 34);

    v8f c00 = {}, c01 = {}, c02 = {}, c03 = {};
    v8f c10 = {}, c11 = {}, c12 = {}, c13 = {};
    float pn0 = 0.0f, pn1 = 0.0f;

    #pragma unroll
    for (int ks = 0; ks < 3; ++ks) {
        v16h a0, a1;
        if (interior) {
            #pragma unroll
            for (int q = 0; q < 3; ++q) {
                int ro = g ? rowoff_u(6 * ks + 3 + q) : rowoff_u(6 * ks + q);
                float msk = g ? rowmsk(ks, 1, q) : 1.0f;
                const float* p = xb + (ro + pb);
                #pragma unroll
                for (int j = 0; j < 5; ++j) {
                    float v0 = p[j]      * msk;
                    float v1 = p[j + 16] * msk;
                    pn0 += v0 * v0;  pn1 += v1 * v1;
                    a0[q * 5 + j] = (_Float16)v0;
                    a1[q * 5 + j] = (_Float16)v1;
                }
            }
        } else {
            #pragma unroll
            for (int q = 0; q < 3; ++q) {
                int ro = g ? rowoff_u(6 * ks + 3 + q) : rowoff_u(6 * ks + q);
                int dy = g ? rowdy_u(6 * ks + 3 + q) : rowdy_u(6 * ks + q);
                int yy = y + dy;
                bool yok = (unsigned)yy < HH;
                #pragma unroll
                for (int j = 0; j < 5; ++j) {
                    int  xx0 = xw + j - 2;
                    bool ok0 = yok && ((unsigned)xx0 < WW);
                    bool ok1 = yok && ((unsigned)(xx0 + 16) < WW);
                    int  ix0 = ok0 ? (ro + pb + j)      : 0;
                    int  ix1 = ok1 ? (ro + pb + j + 16) : 0;
                    float v0 = xb[ix0];  v0 = ok0 ? v0 : 0.0f;
                    float v1 = xb[ix1];  v1 = ok1 ? v1 : 0.0f;
                    pn0 += v0 * v0;  pn1 += v1 * v1;
                    a0[q * 5 + j] = (_Float16)v0;
                    a1[q * 5 + j] = (_Float16)v1;
                }
            }
        }
        a0[15] = (_Float16)0.0f;
        a1[15] = (_Float16)0.0f;

        #pragma unroll
        for (int nt = 0; nt < 4; ++nt) {
            int n = nt * 16 + m;
            const v8h* bp = (const v8h*)&Wh[n * KPAD + ks * 32 + g * 16];
            v8h blo = bp[0], bhi = bp[1];
            v16h bf;
            #pragma unroll
            for (int i2 = 0; i2 < 8; ++i2) { bf[i2] = blo[i2]; bf[8 + i2] = bhi[i2]; }
            switch (nt) {
            case 0: c00 = __builtin_amdgcn_wmma_f32_16x16x32_f16(false, a0, false, bf, (short)0, c00, false, false);
                    c10 = __builtin_amdgcn_wmma_f32_16x16x32_f16(false, a1, false, bf, (short)0, c10, false, false); break;
            case 1: c01 = __builtin_amdgcn_wmma_f32_16x16x32_f16(false, a0, false, bf, (short)0, c01, false, false);
                    c11 = __builtin_amdgcn_wmma_f32_16x16x32_f16(false, a1, false, bf, (short)0, c11, false, false); break;
            case 2: c02 = __builtin_amdgcn_wmma_f32_16x16x32_f16(false, a0, false, bf, (short)0, c02, false, false);
                    c12 = __builtin_amdgcn_wmma_f32_16x16x32_f16(false, a1, false, bf, (short)0, c12, false, false); break;
            default:c03 = __builtin_amdgcn_wmma_f32_16x16x32_f16(false, a0, false, bf, (short)0, c03, false, false);
                    c13 = __builtin_amdgcn_wmma_f32_16x16x32_f16(false, a1, false, bf, (short)0, c13, false, false); break;
            }
        }
    }

    float* dw = dist_s[wave];
    epilogue_half(dw, c00, c01, c02, c03, pn0, g, m, lane, wn_s, out, b, l0);
    epilogue_half(dw, c10, c11, c12, c13, pn1, g, m, lane, wn_s, out, b, l0 + 16);
}

extern "C" void kernel_launch(void* const* d_in, const int* in_sizes, int n_in,
                              void* d_out, int out_size, void* d_ws, size_t ws_size,
                              hipStream_t stream) {
    const float* x = (const float*)d_in[0];   // (32, 3, 224, 224) f32
    const float* w = (const float*)d_in[1];   // (64, 75) f32
    float* out = (float*)d_out;               // (32, 64, 224, 224) f32

    const int blocks = (BATCH * T32_PER_IMG) / WAVES_PER_BLK;   // 6272

    if (ws_size >= WS_TOTAL) {
        _Float16* whp = (_Float16*)d_ws;
        float*    wnp = (float*)((char*)d_ws + WS_WH_BYTES);
        float*    xp  = (float*)((char*)d_ws + XP_OFF);
        rbf_prep<<<1, 256, 0, stream>>>(w, whp, wnp);
        const int padN = BATCH * PIMG;                          // 4,990,464
        rbf_pad<<<(padN + 255) / 256, 256, 0, stream>>>(x, xp);
        rbf_main_padded<<<blocks, 256, 0, stream>>>(xp, whp, wnp, out);
    } else {
        rbf_main_fallback<<<blocks, 256, 0, stream>>>(x, w, out);
    }
}
